// InterDST_LSTMCell_78151224918069
// MI455X (gfx1250) — matrix operand order repaired and checked by reading, not compiled, not run on hardware
//
#include <hip/hip_runtime.h>

typedef __bf16 bf16;
typedef bf16  v16bf __attribute__((ext_vector_type(16)));
typedef bf16  v8bf  __attribute__((ext_vector_type(8)));
typedef float v8f   __attribute__((ext_vector_type(8)));
typedef unsigned int u32x4 __attribute__((ext_vector_type(4)));
typedef int          i32x4 __attribute__((ext_vector_type(4)));
typedef int          i32x8 __attribute__((ext_vector_type(8)));

// ---------------------------------------------------------------------------
// WMMA helpers (CDNA5: D = A(16x32 bf16) * B(32x16 bf16) + C(16x16 f32))
// ---------------------------------------------------------------------------
__device__ __forceinline__ v8f wmma_bf16(v16bf a, v16bf b, v8f c) {
  return __builtin_amdgcn_wmma_f32_16x16x32_bf16(false, a, false, b, (short)0, c,
                                                 false, false);
}

// A-operand load in the contiguous k order used here (B loads match): lane (g = lane>>4) holds
// K in {g*16..g*16+15}; build the v16bf from two 16B loads.
__device__ __forceinline__ v16bf ldA(const bf16* chunk, int g) {
  v8bf lo = *(const v8bf*)(chunk + g * 16);
  v8bf hi = *(const v8bf*)(chunk + g * 16 + 8);
  return __builtin_shufflevector(lo, hi, 0, 1, 2, 3, 4, 5, 6, 7, 8, 9, 10, 11,
                                 12, 13, 14, 15);
}

__device__ __forceinline__ v8f vz8f() {
  v8f z; for (int i = 0; i < 8; ++i) z[i] = 0.f; return z;
}
__device__ __forceinline__ v16bf vz16bf() {
  v16bf z; for (int i = 0; i < 16; ++i) z[i] = (bf16)0.f; return z;
}
__device__ __forceinline__ float sigf(float v) { return 1.f / (1.f + __expf(-v)); }

// ---------------------------------------------------------------------------
// Tensor Data Mover: 1-D tile (nel bf16 elements) global -> LDS.
// D# built per CDNA5 ISA; issued by one wave,
// completion tracked with TENSORcnt.
// ---------------------------------------------------------------------------
__device__ __forceinline__ void tdm_load_1d(unsigned ldsOff,
                                            unsigned long long ga,
                                            unsigned nel) {
  u32x4 g0;
  g0[0] = 1u;                                  // count=1 (valid), user mode
  g0[1] = ldsOff;                              // lds_addr (bytes)
  g0[2] = (unsigned)ga;                        // global_addr[31:0]
  g0[3] = (unsigned)(ga >> 32) | (2u << 30);   // global_addr[56:32] | type=2
  i32x8 g1;
  g1[0] = (int)(1u << 16);                     // data_size = 1 -> 2 bytes
  g1[1] = (int)((nel & 0xffffu) << 16);        // tensor_dim0[15:0]
  g1[2] = (int)((nel >> 16) & 0xffffu);        // tensor_dim0[31:16]; dim1=0
  g1[3] = (int)((nel & 0xffffu) << 16);        // tile_dim0 = nel
  g1[4] = 0;                                   // tile_dim1/2 unused
  g1[5] = (int)nel;                            // tensor_dim0_stride lo
  g1[6] = 0;
  g1[7] = 0;
  i32x4 z4; z4[0] = 0; z4[1] = 0; z4[2] = 0; z4[3] = 0;
#if __clang_major__ >= 23
  i32x8 z8; for (int i = 0; i < 8; ++i) z8[i] = 0;
  __builtin_amdgcn_tensor_load_to_lds(g0, g1, z4, z4, z8, 0);
#else
  __builtin_amdgcn_tensor_load_to_lds(g0, g1, z4, z4, 0);
#endif
}

// ---------------------------------------------------------------------------
// Layout / conversion kernels
// ---------------------------------------------------------------------------
// src [nb][C][1024] f32 -> dst [nb][1024][C] bf16
__global__ __launch_bounds__(256) void pack_nhwc_k(const float* __restrict__ src,
                                                   bf16* __restrict__ dst,
                                                   int nb, int C) {
  size_t i = (size_t)blockIdx.x * 256 + threadIdx.x;
  size_t total = (size_t)nb * C * 1024;
  if (i >= total) return;
  int c = (int)(i % C);
  size_t r = i / C;
  int p = (int)(r % 1024);
  int s = (int)(r / 1024);
  dst[i] = (bf16)src[((size_t)s * C + c) * 1024 + p];
}

__global__ __launch_bounds__(256) void cvt_bf16_k(const float* __restrict__ src,
                                                  bf16* __restrict__ dst, size_t n) {
  size_t i = (size_t)blockIdx.x * 256 + threadIdx.x;
  if (i < n) dst[i] = (bf16)src[i];
}

// w [Cout][Cin][KK] f32 -> wp [KK][Cout][Cin] bf16
__global__ __launch_bounds__(256) void pack_w_k(const float* __restrict__ w,
                                                bf16* __restrict__ wp,
                                                int Cout, int Cin, int KK) {
  size_t i = (size_t)blockIdx.x * 256 + threadIdx.x;
  size_t total = (size_t)Cout * Cin * KK;
  if (i >= total) return;
  int ci = (int)(i % Cin);
  size_t r = i / Cin;
  int co = (int)(r % Cout);
  int p = (int)(r / Cout);
  wp[i] = (bf16)w[((size_t)co * Cin + ci) * KK + p];
}

// ---------------------------------------------------------------------------
// Implicit-GEMM convolution, SAME padding, stride 1, NHWC bf16 input,
// weights pre-packed [p][Cout][Cin] bf16, f32 CHW output.
// Block = 256 threads (8 waves). Block tile: 16 cout x 128 pixels.
// Each wave: 16 cout x 16 pixels, K = CIN per kernel tap.
// Weight slab (16 x CIN bf16) staged into LDS by the Tensor Data Mover.
// ---------------------------------------------------------------------------
template <int CIN, int COUT, int KS>
__global__ __launch_bounds__(256) void conv_wmma_k(const bf16* __restrict__ in,
                                                   const bf16* __restrict__ wp,
                                                   const float* __restrict__ bias,
                                                   float* __restrict__ out) {
  __shared__ bf16 wl[16 * CIN] __attribute__((aligned(64)));
  const int tid = threadIdx.x;
  const int lane = tid & 31;
  const int wave = tid >> 5;
  const int g = lane >> 4;
  const int m = lane & 15;

  int bid = blockIdx.x;
  const int pt = bid & 7;            // 8 pixel tiles of 128
  bid >>= 3;
  constexpr int NCT = COUT >> 4;
  const int ct = bid % NCT;
  const int b = bid / NCT;
  const int coutBase = ct << 4;

  const int pix = pt * 128 + wave * 16 + m;   // this lane's output pixel (N)
  const int y = pix >> 5;
  const int x = pix & 31;
  constexpr int R = KS >> 1;
  const unsigned ldsOff = (unsigned)(size_t)(void*)wl;

  v8f acc = vz8f();
  constexpr int TAPS = KS * KS;
  for (int p = 0; p < TAPS; ++p) {
    __syncthreads();                 // protect LDS WAR across taps
    if (wave == 0) {
      const unsigned long long ga =
          (unsigned long long)(const void*)(wp + ((size_t)p * COUT + coutBase) * CIN);
      tdm_load_1d(ldsOff, ga, 16u * CIN);
      __builtin_amdgcn_s_wait_tensorcnt(0);
    }
    __syncthreads();
    const int dy = p / KS - R;
    const int dx = p % KS - R;
    const int iy = y + dy;
    const int ix = x + dx;
    const bool ok = (iy >= 0) & (iy < 32) & (ix >= 0) & (ix < 32);
    const bf16* bp = in + (size_t)(b * 1024 + iy * 32 + ix) * CIN;
#pragma unroll
    for (int kc = 0; kc < (CIN >> 5); ++kc) {
      v16bf av = ldA(wl + m * CIN + kc * 32, g);
      v16bf bv = vz16bf();
      if (ok) bv = *(const v16bf*)(bp + kc * 32 + g * 16);
      acc = wmma_bf16(av, bv, acc);
    }
  }
#pragma unroll
  for (int i = 0; i < 8; ++i) {
    const int co = coutBase + i + 8 * g;     // M = i + 8*g
    float v = acc[i];
    if (bias) v += bias[co];
    out[((size_t)b * COUT + co) * 1024 + pix] = v;
  }
}

// ---------------------------------------------------------------------------
// Spatial attention (flash-style): Q [1024,64], K=V [4096,64] per batch.
// qn: f_t NHWC bf16; kn: c_hist NHWC bf16 [b][4096][64];
// vc: c_hist CHW bf16 [b][4][64][1024]; out = c_t + attn (f32 CHW).
// Grid = 16 batches * 8 q-blocks; block = 256 (8 waves, 16 q-rows each).
// ---------------------------------------------------------------------------
__global__ __launch_bounds__(256) void attn_spatial_k(const bf16* __restrict__ qn,
                                                      const bf16* __restrict__ kn,
                                                      const bf16* __restrict__ vc,
                                                      const float* __restrict__ c_t,
                                                      float* __restrict__ outp) {
  __shared__ bf16 plds[8][16 * 32] __attribute__((aligned(32)));
  const int tid = threadIdx.x;
  const int lane = tid & 31;
  const int wave = tid >> 5;
  const int g = lane >> 4;
  const int m = lane & 15;
  const int b = blockIdx.x >> 3;
  const int qt = blockIdx.x & 7;
  const int qBase = qt * 128 + wave * 16;
  const int qRow = qBase + m;

  const bf16* qrow = qn + ((size_t)b * 1024 + qRow) * 64;
  const v16bf a0 = ldA(qrow, g);        // channels 0..31
  const v16bf a1 = ldA(qrow + 32, g);   // channels 32..63

  float rmax[8], rsum[8];
#pragma unroll
  for (int i = 0; i < 8; ++i) { rmax[i] = -3.0e38f; rsum[i] = 0.f; }
  v8f o[4];
#pragma unroll
  for (int t = 0; t < 4; ++t) o[t] = vz8f();

  bf16* pl = &plds[wave][0];

  for (int kt = 0; kt < 128; ++kt) {
    const int k0 = kt * 32;
    // prefetch next K-tile rows into GL2 (lowers to global_prefetch_b8)
    if (kt < 127) {
      __builtin_prefetch((const void*)(kn + ((size_t)b * 4096 + k0 + 32 + m) * 64), 0, 1);
      __builtin_prefetch((const void*)(kn + ((size_t)b * 4096 + k0 + 48 + m) * 64), 0, 1);
    }
    // ---- scores: S[16q x 32keys], K-dim = 64 channels
    v8f s0 = vz8f(), s1 = vz8f();
    {
      const bf16* kr0 = kn + ((size_t)b * 4096 + k0 + m) * 64;
      s0 = wmma_bf16(a0, *(const v16bf*)(kr0 + g * 16), s0);
      s0 = wmma_bf16(a1, *(const v16bf*)(kr0 + 32 + g * 16), s0);
      const bf16* kr1 = kn + ((size_t)b * 4096 + k0 + 16 + m) * 64;
      s1 = wmma_bf16(a0, *(const v16bf*)(kr1 + g * 16), s1);
      s1 = wmma_bf16(a1, *(const v16bf*)(kr1 + 32 + g * 16), s1);
    }
    // ---- online softmax (row r = i + 8*g lives on the 16 lanes of group g)
    float p0v[8], p1v[8];
#pragma unroll
    for (int i = 0; i < 8; ++i) {
      float cm = fmaxf(s0[i], s1[i]);
      cm = fmaxf(cm, __shfl_xor(cm, 1));
      cm = fmaxf(cm, __shfl_xor(cm, 2));
      cm = fmaxf(cm, __shfl_xor(cm, 4));
      cm = fmaxf(cm, __shfl_xor(cm, 8));
      const float nm = fmaxf(rmax[i], cm);
      const float sc = __expf(rmax[i] - nm);
      p0v[i] = __expf(s0[i] - nm);
      p1v[i] = __expf(s1[i] - nm);
      float ls = p0v[i] + p1v[i];
      ls += __shfl_xor(ls, 1);
      ls += __shfl_xor(ls, 2);
      ls += __shfl_xor(ls, 4);
      ls += __shfl_xor(ls, 8);
      rsum[i] = rsum[i] * sc + ls;
      rmax[i] = nm;
      o[0][i] *= sc; o[1][i] *= sc; o[2][i] *= sc; o[3][i] *= sc;
    }
    // ---- stage P (C/D layout -> LDS row-major), reload in A layout
#pragma unroll
    for (int i = 0; i < 8; ++i) {
      const int r = i + 8 * g;
      pl[r * 32 + m]      = (bf16)p0v[i];
      pl[r * 32 + 16 + m] = (bf16)p1v[i];
    }
    asm volatile("s_wait_dscnt 0" ::: "memory");
    const v16bf pa = ldA(pl + m * 32, g);
    // ---- P * V : V chunk [32 keys x 64 ch], contiguous keys in CHW layout
    const int kb = k0 + g * 16;
    const int lblk = kb >> 10;
    const int pp = kb & 1023;
    const bf16* vbase = vc + (((size_t)b * 4 + lblk) * 64) * 1024 + pp;
    o[0] = wmma_bf16(pa, *(const v16bf*)(vbase + (size_t)(m) * 1024), o[0]);
    o[1] = wmma_bf16(pa, *(const v16bf*)(vbase + (size_t)(16 + m) * 1024), o[1]);
    o[2] = wmma_bf16(pa, *(const v16bf*)(vbase + (size_t)(32 + m) * 1024), o[2]);
    o[3] = wmma_bf16(pa, *(const v16bf*)(vbase + (size_t)(48 + m) * 1024), o[3]);
  }
  // ---- epilogue: normalize, add c_t, store f32 CHW
  float inv[8];
#pragma unroll
  for (int i = 0; i < 8; ++i) inv[i] = 1.f / rsum[i];
#pragma unroll
  for (int t = 0; t < 4; ++t) {
#pragma unroll
    for (int i = 0; i < 8; ++i) {
      const int ch = t * 16 + m;
      const int pix = qBase + i + 8 * g;
      const size_t offo = ((size_t)b * 64 + ch) * 1024 + pix;
      outp[offo] = c_t[offo] + o[t][i] * inv[i];
    }
  }
}

// ---------------------------------------------------------------------------
// Channel attention: scores [64,256] per batch, softmax over 256, P*KV.
// qc: f_t CHW bf16 [b][64][1024]; vc: c_hist CHW bf16 [b][256][1024];
// kn: c_hist NHWC bf16 [b][4096][64]; out = c_t + attn (f32 CHW).
// Grid = 16 (one block per batch), block = 256.
// ---------------------------------------------------------------------------
__global__ __launch_bounds__(256) void attn_channel_k(const bf16* __restrict__ qc,
                                                      const bf16* __restrict__ vc,
                                                      const bf16* __restrict__ kn,
                                                      const float* __restrict__ c_t,
                                                      float* __restrict__ outp) {
  __shared__ bf16 sl[64 * 256] __attribute__((aligned(32)));   // 32 KB
  const int tid = threadIdx.x;
  const int lane = tid & 31;
  const int wave = tid >> 5;
  const int g = lane >> 4;
  const int m = lane & 15;
  const int b = blockIdx.x;

  // ---- phase 1: scores = Q * KV^T   (M=64 q-ch, N=256 kv-ch, K=1024 pix)
  for (int t = wave; t < 64; t += 8) {
    const int Mt = t >> 4, Nt = t & 15;
    v8f acc = vz8f();
    const bf16* arow = qc + ((size_t)b * 64 + Mt * 16 + m) * 1024;
    const bf16* brow = vc + ((size_t)b * 256 + Nt * 16 + m) * 1024;
#pragma unroll 4
    for (int kc = 0; kc < 32; ++kc) {
      v16bf av = ldA(arow + kc * 32, g);
      v16bf bv = *(const v16bf*)(brow + kc * 32 + g * 16);
      acc = wmma_bf16(av, bv, acc);
    }
#pragma unroll
    for (int i = 0; i < 8; ++i)
      sl[(Mt * 16 + i + 8 * g) * 256 + Nt * 16 + m] = (bf16)acc[i];
  }
  __syncthreads();
  // ---- phase 2: row softmax (64 rows, one thread each)
  if (tid < 64) {
    const int r = tid;
    float mx = -3.0e38f;
    for (int j = 0; j < 256; ++j) mx = fmaxf(mx, (float)sl[r * 256 + j]);
    float s = 0.f;
    for (int j = 0; j < 256; ++j) s += __expf((float)sl[r * 256 + j] - mx);
    const float inv = 1.f / s;
    for (int j = 0; j < 256; ++j)
      sl[r * 256 + j] = (bf16)(__expf((float)sl[r * 256 + j] - mx) * inv);
  }
  __syncthreads();
  // ---- phase 3: out = P * KV   (M=64 ch, N=1024 pix, K=256 kv-ch)
  for (int t = wave; t < 256; t += 8) {
    const int Mt = t >> 6, Nt = t & 63;
    v8f acc = vz8f();
    const bf16* arow = sl + (Mt * 16 + m) * 256;
    const int pix = Nt * 16 + m;
#pragma unroll
    for (int kc = 0; kc < 8; ++kc) {
      v16bf av = ldA(arow + kc * 32, g);
      const int q0 = kc * 32 + g * 16;
      const int lblk = q0 >> 6;
      const int c0 = q0 & 63;
      v16bf bv = *(const v16bf*)(kn + (((size_t)b * 4096) + lblk * 1024 + pix) * 64 + c0);
      acc = wmma_bf16(av, bv, acc);
    }
#pragma unroll
    for (int i = 0; i < 8; ++i) {
      const int ch = Mt * 16 + i + 8 * g;
      const size_t offo = ((size_t)b * 64 + ch) * 1024 + pix;
      outp[offo] = c_t[offo] + acc[i];
    }
  }
}

// ---------------------------------------------------------------------------
// LayerNorm stats: per sample b, mean/rstd over N elements -> stats[2b],[2b+1]
// ---------------------------------------------------------------------------
__global__ __launch_bounds__(256) void ln_stats_k(const float* __restrict__ x,
                                                  float* __restrict__ stats, int N) {
  __shared__ float s1[256], s2[256];
  const int b = blockIdx.x;
  const float* p = x + (size_t)b * N;
  float a = 0.f, q = 0.f;
  for (int i = threadIdx.x; i < N; i += 256) { float v = p[i]; a += v; q += v * v; }
  s1[threadIdx.x] = a; s2[threadIdx.x] = q;
  __syncthreads();
  for (int s = 128; s > 0; s >>= 1) {
    if (threadIdx.x < (unsigned)s) {
      s1[threadIdx.x] += s1[threadIdx.x + s];
      s2[threadIdx.x] += s2[threadIdx.x + s];
    }
    __syncthreads();
  }
  if (threadIdx.x == 0) {
    const float mu = s1[0] / N;
    const float var = s2[0] / N - mu * mu;
    stats[2 * b] = mu;
    stats[2 * b + 1] = rsqrtf(var + 1e-5f);
  }
}

// LN apply (+optional ReLU) -> bf16 NHWC.  x f32 [16][64][1024].
__global__ __launch_bounds__(256) void ln_apply_nhwc_k(const float* __restrict__ x,
                                                       const float* __restrict__ stats,
                                                       const float* __restrict__ gg,
                                                       const float* __restrict__ bb,
                                                       bf16* __restrict__ dst, int relu) {
  size_t i = (size_t)blockIdx.x * 256 + threadIdx.x;
  if (i >= (size_t)16 * 64 * 1024) return;
  const int c = (int)(i & 63);
  const size_t r = i >> 6;
  const int p = (int)(r & 1023);
  const int b = (int)(r >> 10);
  const float mu = stats[2 * b], rs = stats[2 * b + 1];
  const size_t ai = (size_t)c * 1024 + p;
  float v = (x[((size_t)b * 64 + c) * 1024 + p] - mu) * rs * gg[ai] + bb[ai];
  if (relu) v = fmaxf(v, 0.f);
  dst[i] = (bf16)v;
}

// ---------------------------------------------------------------------------
// Fused gate math: applies LN to x/h/m concats, emits f_t, i_t*g_t, o_x+o_h,
// and m_new (written straight into d_out's m slot).
// ---------------------------------------------------------------------------
__global__ __launch_bounds__(256) void gates_k(const float* __restrict__ xc,
                                               const float* __restrict__ hc,
                                               const float* __restrict__ mc,
                                               const float* __restrict__ m_t,
                                               const float* __restrict__ sx,
                                               const float* __restrict__ sh,
                                               const float* __restrict__ sm,
                                               const float* __restrict__ gx,
                                               const float* __restrict__ bx,
                                               const float* __restrict__ gh,
                                               const float* __restrict__ bh,
                                               const float* __restrict__ gm,
                                               const float* __restrict__ bm,
                                               float* __restrict__ f_t,
                                               float* __restrict__ ig,
                                               float* __restrict__ oxh,
                                               float* __restrict__ m_out) {
  size_t e = (size_t)blockIdx.x * 256 + threadIdx.x;
  if (e >= (size_t)16 * 64 * 1024) return;
  const int b = (int)(e >> 16);
  const int c = (int)((e >> 10) & 63);
  const int p = (int)(e & 1023);
  const float mxv = sx[2 * b], rxv = sx[2 * b + 1];
  const float mhv = sh[2 * b], rhv = sh[2 * b + 1];
  const float mmv = sm[2 * b], rmv = sm[2 * b + 1];
  auto lnx = [&](int s) -> float {
    const size_t o = ((size_t)b * 448 + s * 64 + c) * 1024 + p;
    const size_t ai = ((size_t)s * 64 + c) * 1024 + p;
    return (xc[o] - mxv) * rxv * gx[ai] + bx[ai];
  };
  auto lnh = [&](int s) -> float {
    const size_t o = ((size_t)b * 256 + s * 64 + c) * 1024 + p;
    const size_t ai = ((size_t)s * 64 + c) * 1024 + p;
    return (hc[o] - mhv) * rhv * gh[ai] + bh[ai];
  };
  auto lnm = [&](int s) -> float {
    const size_t o = ((size_t)b * 192 + s * 64 + c) * 1024 + p;
    const size_t ai = ((size_t)s * 64 + c) * 1024 + p;
    return (mc[o] - mmv) * rmv * gm[ai] + bm[ai];
  };
  const float i_x = lnx(0), f_x = lnx(1), g_x = lnx(2);
  const float i_xp = lnx(3), f_xp = lnx(4), g_xp = lnx(5), o_x = lnx(6);
  const float i_h = lnh(0), f_h = lnh(1), g_h = lnh(2), o_h = lnh(3);
  const float i_m = lnm(0), f_m = lnm(1), g_m = lnm(2);
  const float it = sigf(i_x + i_h);
  const float ft = sigf(f_x + f_h + 1.0f);
  const float gt = tanhf(g_x + g_h);
  f_t[e] = ft;
  ig[e] = it * gt;
  oxh[e] = o_x + o_h;
  const float itp = sigf(i_xp + i_m);
  const float ftp = sigf(f_xp + f_m + 1.0f);
  const float gtp = tanhf(g_xp + g_m);
  m_out[e] = ftp * m_t[e] + itp * gtp;
}

// s_a + c_a -> bf16 NHWC
__global__ __launch_bounds__(256) void sum_pack_nhwc_k(const float* __restrict__ a,
                                                       const float* __restrict__ b2,
                                                       bf16* __restrict__ dst) {
  size_t i = (size_t)blockIdx.x * 256 + threadIdx.x;
  if (i >= (size_t)16 * 64 * 1024) return;
  const int c = (int)(i & 63);
  const size_t r = i >> 6;
  const int p = (int)(r & 1023);
  const int b = (int)(r >> 10);
  const size_t o = ((size_t)b * 64 + c) * 1024 + p;
  dst[i] = (bf16)(a[o] + b2[o]);
}

// c_new = attn + i_t*g_t -> d_out c slot; also build mem NHWC bf16 [b][p][128]
__global__ __launch_bounds__(256) void cnew_mem_k(const float* __restrict__ attn,
                                                  const float* __restrict__ ig,
                                                  const float* __restrict__ m_new,
                                                  float* __restrict__ c_out,
                                                  bf16* __restrict__ mem) {
  size_t e = (size_t)blockIdx.x * 256 + threadIdx.x;
  if (e >= (size_t)16 * 64 * 1024) return;
  const int b = (int)(e >> 16);
  const int c = (int)((e >> 10) & 63);
  const int p = (int)(e & 1023);
  const float cv = attn[e] + ig[e];
  c_out[e] = cv;
  const size_t mo = (((size_t)b * 1024) + p) * 128;
  mem[mo + c] = (bf16)cv;
  mem[mo + 64 + c] = (bf16)m_new[e];
}

// h_new = sigmoid(oxh + LN_o(o_raw)) * tanh(last)
__global__ __launch_bounds__(256) void final_k(const float* __restrict__ o_raw,
                                               const float* __restrict__ st_o,
                                               const float* __restrict__ go,
                                               const float* __restrict__ bo,
                                               const float* __restrict__ oxh,
                                               const float* __restrict__ last,
                                               float* __restrict__ h_out) {
  size_t e = (size_t)blockIdx.x * 256 + threadIdx.x;
  if (e >= (size_t)16 * 64 * 1024) return;
  const int b = (int)(e >> 16);
  const int c = (int)((e >> 10) & 63);
  const int p = (int)(e & 1023);
  const float mu = st_o[2 * b], rs = st_o[2 * b + 1];
  const size_t ai = (size_t)c * 1024 + p;
  const float oln = (o_raw[e] - mu) * rs * go[ai] + bo[ai];
  const float ot = sigf(oxh[e] + oln);
  h_out[e] = ot * tanhf(last[e]);
}

// ---------------------------------------------------------------------------
// Host orchestration
// ---------------------------------------------------------------------------
extern "C" void kernel_launch(void* const* d_in, const int* in_sizes, int n_in,
                              void* d_out, int out_size, void* d_ws, size_t ws_size,
                              hipStream_t stream) {
  (void)in_sizes; (void)n_in; (void)out_size; (void)ws_size;
  const float* x_t = (const float*)d_in[0];
  const float* h_t = (const float*)d_in[1];
  const float* c_t = (const float*)d_in[2];
  const float* c_hist = (const float*)d_in[3];
  const float* m_t = (const float*)d_in[4];
  const float* w_x = (const float*)d_in[5];
  const float* ln_x_g = (const float*)d_in[6];
  const float* ln_x_b = (const float*)d_in[7];
  const float* w_h = (const float*)d_in[8];
  const float* ln_h_g = (const float*)d_in[9];
  const float* ln_h_b = (const float*)d_in[10];
  const float* w_m = (const float*)d_in[11];
  const float* ln_m_g = (const float*)d_in[12];
  const float* ln_m_b = (const float*)d_in[13];
  const float* w_o = (const float*)d_in[14];
  const float* ln_o_g = (const float*)d_in[15];
  const float* ln_o_b = (const float*)d_in[16];
  const float* w_last = (const float*)d_in[17];
  const float* s_norm_g = (const float*)d_in[18];
  const float* s_norm_b = (const float*)d_in[19];
  const float* c_norm_g = (const float*)d_in[20];
  const float* c_norm_b = (const float*)d_in[21];
  const float* ca_w1 = (const float*)d_in[22];
  const float* ca_b1 = (const float*)d_in[23];
  const float* ca_ln_g = (const float*)d_in[24];
  const float* ca_ln_b = (const float*)d_in[25];
  const float* ca_w2 = (const float*)d_in[26];
  const float* ca_b2 = (const float*)d_in[27];
  const float* sa_w1 = (const float*)d_in[28];
  const float* sa_ln_g = (const float*)d_in[29];
  const float* sa_ln_b = (const float*)d_in[30];
  const float* sa_w2 = (const float*)d_in[31];
  const float* attn_w = (const float*)d_in[32];

  const size_t M1 = (size_t)16 * 64 * 1024;   // 1048576
  float* h_out = (float*)d_out;
  float* c_out = h_out + M1;
  float* m_out = c_out + M1;

  char* ws = (char*)d_ws;
  size_t off = 0;
  auto A = [&](size_t bytes) -> char* {
    char* p = ws + off;
    off += (bytes + 255) & ~(size_t)255;
    return p;
  };

  // bf16 buffers
  bf16* xt_n  = (bf16*)A(M1 * 2);
  bf16* ht_n  = (bf16*)A(M1 * 2);
  bf16* mt_n  = (bf16*)A(M1 * 2);
  bf16* ft_n  = (bf16*)A(M1 * 2);
  bf16* ft_c  = (bf16*)A(M1 * 2);
  bf16* ch_n  = (bf16*)A(M1 * 4 * 2);     // [b][4096][64]
  bf16* cv_c  = (bf16*)A(M1 * 4 * 2);     // [b][256][1024]
  bf16* spn   = (bf16*)A(M1 * 2);
  bf16* chn   = (bf16*)A(M1 * 2);
  bf16* sa1r  = (bf16*)A(M1 * 2);
  bf16* ca1r  = (bf16*)A(M1 * 2);
  bf16* sc_n  = (bf16*)A(M1 * 2);
  bf16* mem_n = (bf16*)A(M1 * 2 * 2);     // [b][1024][128]
  bf16* wx_p  = (bf16*)A((size_t)25 * 448 * 64 * 2);
  bf16* wh_p  = (bf16*)A((size_t)25 * 256 * 64 * 2);
  bf16* wm_p  = (bf16*)A((size_t)25 * 192 * 64 * 2);
  bf16* wo_p  = (bf16*)A((size_t)25 * 64 * 128 * 2);
  bf16* wl_p  = (bf16*)A((size_t)64 * 128 * 2);
  bf16* sa1_p = (bf16*)A((size_t)25 * 64 * 64 * 2);
  bf16* ca1_p = (bf16*)A((size_t)25 * 64 * 64 * 2);
  bf16* sa2_p = (bf16*)A((size_t)64 * 64 * 2);
  bf16* ca2_p = (bf16*)A((size_t)64 * 64 * 2);
  bf16* aw_p  = (bf16*)A((size_t)64 * 64 * 2);
  // f32 buffers
  float* x_cat  = (float*)A((size_t)16 * 448 * 1024 * 4);
  float* h_cat  = (float*)A((size_t)16 * 256 * 1024 * 4);
  float* m_cat  = (float*)A((size_t)16 * 192 * 1024 * 4);
  float* f_t    = (float*)A(M1 * 4);
  float* ig     = (float*)A(M1 * 4);
  float* oxh    = (float*)A(M1 * 4);
  float* sp_raw = (float*)A(M1 * 4);
  float* ch_raw = (float*)A(M1 * 4);
  float* sa1_o  = (float*)A(M1 * 4);
  float* ca1_o  = (float*)A(M1 * 4);
  float* s_a    = (float*)A(M1 * 4);
  float* c_a    = (float*)A(M1 * 4);
  float* attn_o = (float*)A(M1 * 4);
  float* o_raw  = (float*)A(M1 * 4);
  float* lastb  = (float*)A(M1 * 4);
  float* stats  = (float*)A(8 * 32 * 4);
  float* st_x = stats, *st_h = stats + 32, *st_m = stats + 64;
  float* st_sp = stats + 96, *st_ch = stats + 128;
  float* st_sa = stats + 160, *st_ca = stats + 192, *st_o = stats + 224;

  auto blocks = [](size_t n) { return (unsigned)((n + 255) / 256); };

  // --- stage 0: layout conversions -------------------------------------
  pack_nhwc_k<<<blocks(M1), 256, 0, stream>>>(x_t, xt_n, 16, 64);
  pack_nhwc_k<<<blocks(M1), 256, 0, stream>>>(h_t, ht_n, 16, 64);
  pack_nhwc_k<<<blocks(M1), 256, 0, stream>>>(m_t, mt_n, 16, 64);
  pack_nhwc_k<<<blocks(M1 * 4), 256, 0, stream>>>(c_hist, ch_n, 64, 64);
  cvt_bf16_k<<<blocks(M1 * 4), 256, 0, stream>>>(c_hist, cv_c, M1 * 4);
  pack_w_k<<<blocks((size_t)25 * 448 * 64), 256, 0, stream>>>(w_x, wx_p, 448, 64, 25);
  pack_w_k<<<blocks((size_t)25 * 256 * 64), 256, 0, stream>>>(w_h, wh_p, 256, 64, 25);
  pack_w_k<<<blocks((size_t)25 * 192 * 64), 256, 0, stream>>>(w_m, wm_p, 192, 64, 25);
  pack_w_k<<<blocks((size_t)25 * 64 * 128), 256, 0, stream>>>(w_o, wo_p, 64, 128, 25);
  pack_w_k<<<blocks((size_t)64 * 128), 256, 0, stream>>>(w_last, wl_p, 64, 128, 1);
  pack_w_k<<<blocks((size_t)25 * 64 * 64), 256, 0, stream>>>(sa_w1, sa1_p, 64, 64, 25);
  pack_w_k<<<blocks((size_t)25 * 64 * 64), 256, 0, stream>>>(ca_w1, ca1_p, 64, 64, 25);
  pack_w_k<<<blocks((size_t)64 * 64), 256, 0, stream>>>(sa_w2, sa2_p, 64, 64, 1);
  pack_w_k<<<blocks((size_t)64 * 64), 256, 0, stream>>>(ca_w2, ca2_p, 64, 64, 1);
  pack_w_k<<<blocks((size_t)64 * 64), 256, 0, stream>>>(attn_w, aw_p, 64, 64, 1);

  // --- stage 1: big convs + gate LNs -----------------------------------
  conv_wmma_k<64, 448, 5><<<dim3(16 * 28 * 8), 256, 0, stream>>>(xt_n, wx_p, nullptr, x_cat);
  conv_wmma_k<64, 256, 5><<<dim3(16 * 16 * 8), 256, 0, stream>>>(ht_n, wh_p, nullptr, h_cat);
  conv_wmma_k<64, 192, 5><<<dim3(16 * 12 * 8), 256, 0, stream>>>(mt_n, wm_p, nullptr, m_cat);
  ln_stats_k<<<16, 256, 0, stream>>>(x_cat, st_x, 448 * 1024);
  ln_stats_k<<<16, 256, 0, stream>>>(h_cat, st_h, 256 * 1024);
  ln_stats_k<<<16, 256, 0, stream>>>(m_cat, st_m, 192 * 1024);
  gates_k<<<blocks(M1), 256, 0, stream>>>(x_cat, h_cat, m_cat, m_t, st_x, st_h, st_m,
                                          ln_x_g, ln_x_b, ln_h_g, ln_h_b, ln_m_g,
                                          ln_m_b, f_t, ig, oxh, m_out);
  pack_nhwc_k<<<blocks(M1), 256, 0, stream>>>(f_t, ft_n, 16, 64);
  cvt_bf16_k<<<blocks(M1), 256, 0, stream>>>(f_t, ft_c, M1);

  // --- stage 2: attentions ----------------------------------------------
  attn_spatial_k<<<128, 256, 0, stream>>>(ft_n, ch_n, cv_c, c_t, sp_raw);
  attn_channel_k<<<16, 256, 0, stream>>>(ft_c, cv_c, ch_n, c_t, ch_raw);
  ln_stats_k<<<16, 256, 0, stream>>>(sp_raw, st_sp, 65536);
  ln_apply_nhwc_k<<<blocks(M1), 256, 0, stream>>>(sp_raw, st_sp, s_norm_g, s_norm_b, spn, 0);
  ln_stats_k<<<16, 256, 0, stream>>>(ch_raw, st_ch, 65536);
  ln_apply_nhwc_k<<<blocks(M1), 256, 0, stream>>>(ch_raw, st_ch, c_norm_g, c_norm_b, chn, 0);

  // --- stage 3: attention-fusion conv branches --------------------------
  conv_wmma_k<64, 64, 5><<<dim3(16 * 4 * 8), 256, 0, stream>>>(spn, sa1_p, nullptr, sa1_o);
  ln_stats_k<<<16, 256, 0, stream>>>(sa1_o, st_sa, 65536);
  ln_apply_nhwc_k<<<blocks(M1), 256, 0, stream>>>(sa1_o, st_sa, sa_ln_g, sa_ln_b, sa1r, 1);
  conv_wmma_k<64, 64, 1><<<dim3(16 * 4 * 8), 256, 0, stream>>>(sa1r, sa2_p, nullptr, s_a);
  conv_wmma_k<64, 64, 5><<<dim3(16 * 4 * 8), 256, 0, stream>>>(chn, ca1_p, ca_b1, ca1_o);
  ln_stats_k<<<16, 256, 0, stream>>>(ca1_o, st_ca, 65536);
  ln_apply_nhwc_k<<<blocks(M1), 256, 0, stream>>>(ca1_o, st_ca, ca_ln_g, ca_ln_b, ca1r, 1);
  conv_wmma_k<64, 64, 1><<<dim3(16 * 4 * 8), 256, 0, stream>>>(ca1r, ca2_p, ca_b2, c_a);
  sum_pack_nhwc_k<<<blocks(M1), 256, 0, stream>>>(s_a, c_a, sc_n);
  conv_wmma_k<64, 64, 1><<<dim3(16 * 4 * 8), 256, 0, stream>>>(sc_n, aw_p, nullptr, attn_o);

  // --- stage 4: c_new, mem, output gate ---------------------------------
  cnew_mem_k<<<blocks(M1), 256, 0, stream>>>(attn_o, ig, m_out, c_out, mem_n);
  conv_wmma_k<128, 64, 5><<<dim3(16 * 4 * 8), 256, 0, stream>>>(mem_n, wo_p, nullptr, o_raw);
  ln_stats_k<<<16, 256, 0, stream>>>(o_raw, st_o, 65536);
  conv_wmma_k<128, 64, 1><<<dim3(16 * 4 * 8), 256, 0, stream>>>(mem_n, wl_p, nullptr, lastb);
  final_k<<<blocks(M1), 256, 0, stream>>>(o_raw, st_o, ln_o_g, ln_o_b, oxh, lastb, h_out);
}
